// HybridSurvivalQ_v1_9509057593682
// MI455X (gfx1250) — matrix operand order repaired and checked by reading, not compiled, not run on hardware
//
#include <hip/hip_runtime.h>
#include <hip/hip_bf16.h>

// CDNA5 / gfx1250 fused hybrid quantum-MLP kernel.
//
// q_out[b] = prod_{q=1..6} cos(x_q[b,q] + q_params[q])   (analytic collapse of
//            the product state + CNOT ring + <Z0> measurement: final bit0 of
//            the CNOT-ring permutation is a1^...^a6, so <Z0> factorizes)
// out = relu(relu([q_out, x_c] @ W1 + b1) @ W2 + b2) @ W3 + b3
//
// Layer-1 GEMM via v_wmma_f32_16x16x32_f16 (K=64 over x_c; the q_out column
// is a rank-1 VALU update folded into the accumulator init together with b1).
// Layer-2 GEMM is one more WMMA after an LDS D->A layout bounce. Layer 3 is a
// shfl_xor cross-lane reduction. Each wave loops over TILES_PER_WAVE row
// tiles so the weight-fragment setup is amortized.

typedef __attribute__((ext_vector_type(16))) _Float16 v16h;
typedef __attribute__((ext_vector_type(8)))  _Float16 v8h;
typedef __attribute__((ext_vector_type(8)))  float    v8f;

#define WAVES_PER_BLOCK 8
#define ROWS_PER_WAVE   16
#define ROWS_PER_BLOCK  (WAVES_PER_BLOCK * ROWS_PER_WAVE)
#define TILES_PER_WAVE  4
#define ROWS_PER_BLOCK_TOTAL (ROWS_PER_BLOCK * TILES_PER_WAVE)

__global__ __launch_bounds__(256)
void hybridq_fused_kernel(const float* __restrict__ x_q,
                          const float* __restrict__ x_c,
                          const float* __restrict__ qp,
                          const float* __restrict__ W1,   // (65, 32)
                          const float* __restrict__ b1,   // (32,)
                          const float* __restrict__ W2,   // (32, 16)
                          const float* __restrict__ b2,   // (16,)
                          const float* __restrict__ W3,   // (16, 1)
                          const float* __restrict__ b3,   // (1,)
                          float* __restrict__ out, int B)
{
    const int lane = threadIdx.x & 31;
    const int wave = threadIdx.x >> 5;
    const int half = (lane >> 4) & 1;   // which 16-lane half of the wave
    const int lm   = lane & 15;         // lane index within half

    // -------- weight fragments (f16), loaded once per wave (L2-resident) ----
    // B1: W1 rows 1..64 (K=64) as two 16x16x32 B-fragments per N-tile.
    // B layout (16-bit, 32x16): h<8 -> K = half*8 + h, h>=8 -> K = 16 + half*8 + (h-8), N = lm.
    v16h b1frag[2][2];
    #pragma unroll
    for (int nt = 0; nt < 2; ++nt) {
        #pragma unroll
        for (int s = 0; s < 2; ++s) {
            const int N  = nt * 16 + lm;
            const int kb = half * 8;
            v16h f;
            #pragma unroll
            for (int h = 0; h < 16; ++h) {
                const int K = s * 32 + (h < 8 ? kb + h : 16 + kb + (h - 8));          // 0..63
                f[h] = (_Float16)W1[(size_t)(1 + K) * 32 + N];
            }
            b1frag[nt][s] = f;
        }
    }
    // B2: W2 (32x16), one B-fragment.
    v16h b2frag;
    {
        const int kb = half * 8;
        #pragma unroll
        for (int h = 0; h < 16; ++h)
            b2frag[h] = (_Float16)W2[(size_t)(h < 8 ? kb + h : 16 + kb + (h - 8)) * 16 + lm];
    }
    float w1row0[2], bias1[2];
    w1row0[0] = W1[lm];        w1row0[1] = W1[16 + lm];
    bias1[0]  = b1[lm];        bias1[1]  = b1[16 + lm];
    const float bias2 = b2[lm];
    const float w3    = W3[lm];
    const float bias3 = b3[0];
    // q_params are wave-uniform; pull the 6 used values up front.
    float qpl[6];
    #pragma unroll
    for (int q = 0; q < 6; ++q) qpl[q] = qp[1 + q];

    __shared__ alignas(16) _Float16 h1t[WAVES_PER_BLOCK][16][40]; // stride 40 -> 16B-aligned rows

    const int blockrow0 = blockIdx.x * ROWS_PER_BLOCK_TOTAL;

    for (int t = 0; t < TILES_PER_WAVE; ++t) {
        const int rowbase = blockrow0 + t * ROWS_PER_BLOCK + wave * ROWS_PER_WAVE;
        int r = rowbase + lm; if (r >= B) r = B - 1;

        // Prefetch next tile's x_c row into L2 (global_prefetch_b8).
        if (t + 1 < TILES_PER_WAVE) {
            int rn = rowbase + ROWS_PER_BLOCK + lm; if (rn >= B) rn = B - 1;
            __builtin_prefetch(x_c + (size_t)rn * 64, 0, 1);
        }

        // ---- analytic quantum expectation: row rowbase+lm ------------------
        float qval;
        {
            const float* xr = x_q + (size_t)r * 7;
            float p = 1.0f;
            #pragma unroll
            for (int q = 0; q < 6; ++q)
                p *= __cosf(xr[1 + q] + qpl[q]);   // v_cos_f32 fast path
            qval = p;
        }

        // ---- layer 1: C = x_c(16x64) @ W1[1:,:] + (b1 + q_out * W1[0,:]) ---
        v8f c1[2];
        #pragma unroll
        for (int nt = 0; nt < 2; ++nt) {
            v8f c;
            #pragma unroll
            for (int i = 0; i < 8; ++i) {
                const int m = i + 8 * half;              // row within tile
                const float qm = __shfl(qval, m, 32);    // q_out of row m
                c[i] = bias1[nt] + qm * w1row0[nt];
            }
            c1[nt] = c;
        }

        // A fragments straight from x_c (row-major).
        // A layout (16-bit, 16x32): row = lm; h<8 -> K=kb+h, h>=8 -> K=16+kb+(h-8), kb=8*half.
        {
            const float* xrow = x_c + (size_t)r * 64;
            const int kb = half * 8;
            #pragma unroll
            for (int s = 0; s < 2; ++s) {
                v16h a;
                #pragma unroll
                for (int h = 0; h < 8; ++h) a[h]     = (_Float16)xrow[s * 32 + kb + h];
                #pragma unroll
                for (int h = 0; h < 8; ++h) a[8 + h] = (_Float16)xrow[s * 32 + 16 + kb + h];
                c1[0] = __builtin_amdgcn_wmma_f32_16x16x32_f16(false, a, false, b1frag[0][s],
                                                               (short)0, c1[0], false, false);
                c1[1] = __builtin_amdgcn_wmma_f32_16x16x32_f16(false, a, false, b1frag[1][s],
                                                               (short)0, c1[1], false, false);
            }
        }

        // ---- relu + D-layout -> A-layout bounce through LDS ----------------
        #pragma unroll
        for (int nt = 0; nt < 2; ++nt) {
            #pragma unroll
            for (int i = 0; i < 8; ++i) {
                const int m = i + 8 * half;
                const float v = c1[nt][i];
                h1t[wave][m][nt * 16 + lm] = (_Float16)(v > 0.0f ? v : 0.0f);
            }
        }
        __syncthreads();   // order write -> read (uniform trip count in block)

        v16h a2;
        {
            const _Float16* row = &h1t[wave][lm][0];
            const int kb = half * 8;
            const v8h lo = *(const v8h*)(row + kb);        // K = kb .. kb+7
            const v8h hi = *(const v8h*)(row + 16 + kb);   // K = 16+kb .. 16+kb+7
            #pragma unroll
            for (int h = 0; h < 8; ++h) { a2[h] = lo[h]; a2[8 + h] = hi[h]; }
        }

        // ---- layer 2: 16x32 @ 32x16 + b2 -----------------------------------
        v8f c2;
        #pragma unroll
        for (int i = 0; i < 8; ++i) c2[i] = bias2;
        c2 = __builtin_amdgcn_wmma_f32_16x16x32_f16(false, a2, false, b2frag,
                                                    (short)0, c2, false, false);

        // ---- layer 3: relu -> dot with W3 (cross-lane reduce) --------------
        v8f p;
        #pragma unroll
        for (int i = 0; i < 8; ++i) {
            float v = c2[i];
            v = v > 0.0f ? v : 0.0f;
            p[i] = v * w3;                    // this lane's column n = lm
        }
        #pragma unroll
        for (int mask = 1; mask < 16; mask <<= 1) {
            #pragma unroll
            for (int i = 0; i < 8; ++i)
                p[i] += __shfl_xor(p[i], mask, 32);
        }
        if (lm == 0) {
            const int rb = rowbase + 8 * half;            // rows rb .. rb+7
            #pragma unroll
            for (int i = 0; i < 8; ++i) {
                const int rr = rb + i;
                if (rr < B) out[rr] = p[i] + bias3;
            }
        }
        __syncthreads();   // order read -> next iteration's writes
    }
}

extern "C" void kernel_launch(void* const* d_in, const int* in_sizes, int n_in,
                              void* d_out, int out_size, void* d_ws, size_t ws_size,
                              hipStream_t stream) {
    (void)n_in; (void)d_ws; (void)ws_size; (void)out_size;
    const float* x_q = (const float*)d_in[0];
    const float* x_c = (const float*)d_in[1];
    const float* qp  = (const float*)d_in[2];
    const float* W1  = (const float*)d_in[3];
    const float* b1  = (const float*)d_in[4];
    const float* W2  = (const float*)d_in[5];
    const float* b2  = (const float*)d_in[6];
    const float* W3  = (const float*)d_in[7];
    const float* b3  = (const float*)d_in[8];
    float* out = (float*)d_out;

    const int B = in_sizes[0] / 7;   // x_q is (B, 7)
    const int grid = (B + ROWS_PER_BLOCK_TOTAL - 1) / ROWS_PER_BLOCK_TOTAL;
    hipLaunchKernelGGL(hybridq_fused_kernel, dim3(grid), dim3(256), 0, stream,
                       x_q, x_c, qp, W1, b1, W2, b2, W3, b3, out, B);
}
